// HCLT_55422257988076
// MI455X (gfx1250) — matrix operand order repaired and checked by reading, not compiled, not run on hardware
//
#include <hip/hip_runtime.h>
#include <hip/hip_bf16.h>
#include <math.h>

typedef _Float16 v16h __attribute__((ext_vector_type(16)));
typedef _Float16 v8h  __attribute__((ext_vector_type(8)));
typedef float    v8f  __attribute__((ext_vector_type(8)));

#define MDIM 256                       // latent states == categories == 256
#define KTOT 65536                     // product nodes p = k*256 + m
#define PQ_ELEMS (MDIM * KTOT)         // 16,777,216 elements per matrix (one var)

// ---------------------------------------------------------------------------
// Kernel 1: P[p,c] = sqrt(w_k) * W[v,k,m,c], split into f16 hi + (lo * 2^11),
// stored TRANSPOSED as Pt[c][p] so the GEMM reads contiguous K-rows.
// 64(p) x 64(c) tiles staged through LDS (stride 66 halves -> conflict-free).
// ---------------------------------------------------------------------------
__global__ void hclt_convert(const float* __restrict__ W,
                             const float* __restrict__ w_sum,
                             _Float16* __restrict__ dstHi,
                             _Float16* __restrict__ dstLo,
                             int doLo)
{
    __shared__ _Float16 sHi[64][66];
    __shared__ _Float16 sLo[64][66];
    const int v  = blockIdx.z;
    const int p0 = blockIdx.x * 64;
    const int c0 = blockIdx.y * 64;
    const float*    __restrict__ src   = W     + (size_t)v * PQ_ELEMS;
    _Float16*       __restrict__ outHi = dstHi + (size_t)v * PQ_ELEMS;
    _Float16*       __restrict__ outLo = dstLo + (size_t)v * PQ_ELEMS;
    const int t = threadIdx.x;

    #pragma unroll
    for (int i = 0; i < 16; ++i) {               // coalesced read along c
        int idx = i * 256 + t;
        int pp  = idx >> 6;
        int cc  = idx & 63;
        int p   = p0 + pp;
        float s   = sqrtf(w_sum[p >> 8]);        // k = p / 256
        float val = src[(size_t)p * MDIM + (c0 + cc)] * s;
        _Float16 h = (_Float16)val;
        sHi[pp][cc] = h;
        if (doLo) sLo[pp][cc] = (_Float16)((val - (float)h) * 2048.0f);
    }
    __syncthreads();
    #pragma unroll
    for (int i = 0; i < 16; ++i) {               // coalesced write along p
        int idx = i * 256 + t;
        int cc  = idx >> 6;
        int pp  = idx & 63;
        size_t o = (size_t)(c0 + cc) * KTOT + (p0 + pp);
        outHi[o] = sHi[pp][cc];
        if (doLo) outLo[o] = sLo[pp][cc];
    }
}

// ---------------------------------------------------------------------------
// Kernel 2: A += scale * Pt(rows) x Qt(rows)  via v_wmma_f32_16x16x32_f16.
// 2x4 register blocking: each wave owns a 32(row) x 64(col) output block =
// 8 accumulator tiles, reusing 2 A-fragments x 4 B-fragments per K-step
// (traffic factor (2+4)/(2*4) = 0.75 vs 2.0 for 1x1). Split-K across
// blockIdx.x, reduced with global_atomic_add_f32.
// A-fragment (16x32 f16, contiguous K): lane l -> row r=l&15, halves at
//   K = g*16 + 0..7  and  K = g*16 + 8..15,  g = l>>4.
// B-fragment (32x16 f16): lane l -> col n=l&15, halves at K = g*16 + 0..15.
// C/D (16x16 f32): lane l -> N = l&15, VGPR j -> M = j + 8*g.
// ---------------------------------------------------------------------------
__global__ void hclt_gemm(const _Float16* __restrict__ Pt,
                          const _Float16* __restrict__ Qt,
                          float* __restrict__ Acc,
                          float scale, int kPerChunk)
{
    const int lane = threadIdx.x & 31;
    const int wave = threadIdx.x >> 5;
    const int wt   = blockIdx.y * 8 + wave;      // 0..31 wave-tiles
    const int c0   = (wt >> 2) * 32;             // row base   (8 groups of 32)
    const int c1   = (wt & 3) * 64;              // col base   (4 groups of 64)
    const int r    = lane & 15;
    const int g    = lane >> 4;
    const int kBase = blockIdx.x * kPerChunk;

    const _Float16* ap[2];
    const _Float16* bp[4];
    #pragma unroll
    for (int i = 0; i < 2; ++i)
        ap[i] = Pt + (size_t)(c0 + i * 16 + r) * KTOT + kBase + g * 16;
    #pragma unroll
    for (int j = 0; j < 4; ++j)
        bp[j] = Qt + (size_t)(c1 + j * 16 + r) * KTOT + kBase + g * 16;

    v8f acc[2][4];
    #pragma unroll
    for (int i = 0; i < 2; ++i)
        #pragma unroll
        for (int j = 0; j < 4; ++j)
            acc[i][j] = (v8f){};

    for (int k = 0; k < kPerChunk; k += 32) {
        v16h a[2], b[4];
        #pragma unroll
        for (int i = 0; i < 2; ++i) {
            v8h a0 = *(const v8h*)(ap[i] + k);        // K = g*16 .. +7
            v8h a1 = *(const v8h*)(ap[i] + k + 8);    // K = g*16+8 .. +15
            #pragma unroll
            for (int e = 0; e < 8; ++e) { a[i][e] = a0[e]; a[i][e + 8] = a1[e]; }
        }
        #pragma unroll
        for (int j = 0; j < 4; ++j)
            b[j] = *(const v16h*)(bp[j] + k);         // K = g*16 .. +15
        #pragma unroll
        for (int i = 0; i < 2; ++i)
            #pragma unroll
            for (int j = 0; j < 4; ++j)
                acc[i][j] = __builtin_amdgcn_wmma_f32_16x16x32_f16(
                                false, a[i], false, b[j],
                                (short)0, acc[i][j], false, false);
        __builtin_prefetch(ap[0] + k + 1024, 0, 1);
        __builtin_prefetch(ap[1] + k + 1024, 0, 1);
        __builtin_prefetch(bp[0] + k + 1024, 0, 1);
        __builtin_prefetch(bp[1] + k + 1024, 0, 1);
        __builtin_prefetch(bp[2] + k + 1024, 0, 1);
        __builtin_prefetch(bp[3] + k + 1024, 0, 1);
    }

    #pragma unroll
    for (int i = 0; i < 2; ++i) {
        const int m0 = c0 + i * 16 + g * 8;
        #pragma unroll
        for (int j = 0; j < 4; ++j) {
            const int n = c1 + j * 16 + r;
            #pragma unroll
            for (int e = 0; e < 8; ++e)
                atomicAdd(&Acc[(size_t)(m0 + e) * MDIM + n], acc[i][j][e] * scale);
        }
    }
}

// ---------------------------------------------------------------------------
// Kernel 3: out[b] = log(A[x0[b], x1[b]])
// ---------------------------------------------------------------------------
__global__ void hclt_root(const float* __restrict__ Acc,
                          const int* __restrict__ x,
                          float* __restrict__ out, int nb)
{
    int b = blockIdx.x * blockDim.x + threadIdx.x;
    if (b < nb) {
        int c0 = x[2 * b + 0];
        int c1 = x[2 * b + 1];
        out[b] = logf(Acc[(size_t)c0 * MDIM + c1]);
    }
}

extern "C" void kernel_launch(void* const* d_in, const int* in_sizes, int n_in,
                              void* d_out, int out_size, void* d_ws, size_t ws_size,
                              hipStream_t stream)
{
    (void)in_sizes; (void)n_in;
    const int*   x    = (const int*)d_in[0];    // [B,2] int32
    const float* W    = (const float*)d_in[1];  // [2,256,256,256] f32
    const float* wsum = (const float*)d_in[2];  // [256] f32
    float* out = (float*)d_out;

    char* ws = (char*)d_ws;
    const size_t A_BYTES  = (size_t)MDIM * MDIM * sizeof(float);   // 256 KB
    const size_t PM_BYTES = (size_t)PQ_ELEMS * 2;                  // 32 MB per matrix
    const size_t HI_BYTES = 2 * PM_BYTES;                          // Pt_hi + Qt_hi

    float*    A  = (float*)ws;
    _Float16* hi = (_Float16*)(ws + A_BYTES);
    const int split = (ws_size >= A_BYTES + 2 * HI_BYTES) ? 1 : 0; // room for lo pair?
    _Float16* lo = split ? (_Float16*)(ws + A_BYTES + HI_BYTES) : hi;

    hipMemsetAsync(A, 0, A_BYTES, stream);

    // Convert + transpose both variables (z dim selects var 0/1).
    hclt_convert<<<dim3(KTOT / 64, MDIM / 64, 2), 256, 0, stream>>>(W, wsum, hi, lo, split);

    _Float16* PtHi = hi;
    _Float16* QtHi = (_Float16*)((char*)hi + PM_BYTES);
    _Float16* PtLo = lo;
    _Float16* QtLo = (_Float16*)((char*)lo + PM_BYTES);

    const int KCH = 1024;                 // 64 split-K chunks, 32 K-steps each
    dim3 gg(KTOT / KCH, 4);               // 4 tile-groups x 8 waves = 32 wave-tiles
    hclt_gemm<<<gg, 256, 0, stream>>>(PtHi, QtHi, A, 1.0f, KCH);
    if (split) {
        const float s = 1.0f / 2048.0f;   // undo lo pre-scale
        hclt_gemm<<<gg, 256, 0, stream>>>(PtHi, QtLo, A, s, KCH);
        hclt_gemm<<<gg, 256, 0, stream>>>(PtLo, QtHi, A, s, KCH);
    }

    hclt_root<<<dim3((out_size + 255) / 256), 256, 0, stream>>>(A, x, out, out_size);
}
